// VolFormer_44452911513860
// MI455X (gfx1250) — matrix operand order repaired and checked by reading, not compiled, not run on hardware
//
#include <hip/hip_runtime.h>

// ---------------- problem constants ----------------
#define NL   3
#define D    128
#define H    4
#define DH   32
#define DFF  512
#define DIN  64
#define BB   2
#define LL   2048
#define SS   (LL + 1)          // 2049 tokens incl CLS
#define MM   (BB * SS)         // 4098 rows total

typedef __attribute__((ext_vector_type(16))) _Float16 v16h;
typedef __attribute__((ext_vector_type(8)))  float    v8f;
typedef __attribute__((ext_vector_type(4)))  unsigned v4u;
typedef __attribute__((ext_vector_type(8)))  int      v8i;
typedef __attribute__((ext_vector_type(4)))  int      v4i;

// ---------------- WMMA fragment loaders ----------------
// A-matrix 16x32 f16 layout: lane m = lane%16 (both halves),
//   half=0 holds K=0..15 contiguously, half=1 holds K=16..31 (matches B)
__device__ __forceinline__ v16h load_fragA(const _Float16* base, int row,
                                           int stride, int k0, int half_) {
  const _Float16* p = base + (size_t)row * stride + k0 + half_ * 16;
  v16h a;
  reinterpret_cast<uint4*>(&a)[0] = *reinterpret_cast<const uint4*>(p);
  reinterpret_cast<uint4*>(&a)[1] = *reinterpret_cast<const uint4*>(p + 8);
  return a;
}

// B-matrix 32x16 f16 fragment (contiguous k order, same as A): lane n = lane%16,
//   half=0 holds K=0..15 contiguously, half=1 holds K=16..31.
__device__ __forceinline__ v16h load_fragB(const _Float16* base, int row,
                                           int stride, int k0, int half_) {
  const _Float16* p = base + (size_t)row * stride + k0 + half_ * 16;
  v16h b;
  reinterpret_cast<uint4*>(&b)[0] = reinterpret_cast<const uint4*>(p)[0];
  reinterpret_cast<uint4*>(&b)[1] = reinterpret_cast<const uint4*>(p)[1];
  return b;
}

__device__ __forceinline__ v8f wmma_f16(v16h a, v16h b, v8f c) {
  return __builtin_amdgcn_wmma_f32_16x16x32_f16(false, a, false, b,
                                                (short)0, c, false, false);
}

// ---------------- f32 -> f16 convert ----------------
__global__ __launch_bounds__(256) void k_cvt_f16(const float* __restrict__ s,
                                                 _Float16* __restrict__ d, int n) {
  int i = blockIdx.x * 256 + threadIdx.x;
  if (i < n) d[i] = (_Float16)s[i];
}

// ---------------- embedding ----------------
__global__ __launch_bounds__(256)
void k_embed(const float* __restrict__ x, const float* __restrict__ ew,
             const float* __restrict__ eb, const float* __restrict__ pos,
             const float* __restrict__ cls, float* __restrict__ h) {
  int idx = blockIdx.x * 256 + threadIdx.x;
  if (idx >= MM * D) return;
  int d   = idx & (D - 1);
  int tok = idx >> 7;
  int b   = tok / SS;
  int s   = tok - b * SS;
  float v;
  if (s == 0) {
    v = cls[d] + pos[d];
  } else {
    int l = s - 1;
    const float* xr = x + ((size_t)b * LL + l) * DIN;
    const float* wr = ew + (size_t)d * DIN;
    float acc = 0.f;
    #pragma unroll 8
    for (int k = 0; k < DIN; ++k) acc += xr[k] * wr[k];
    v = acc + eb[d] + pos[(size_t)s * D + d];
  }
  h[idx] = v;
}

// ---------------- LayerNorm -> f16 (1 wave / token) ----------------
__global__ __launch_bounds__(128)
void k_ln_f16(const float* __restrict__ h, const float* __restrict__ sc,
              const float* __restrict__ bi, _Float16* __restrict__ out, int M) {
  int wave = threadIdx.x >> 5, lane = threadIdx.x & 31;
  int tok = blockIdx.x * 4 + wave;
  if (tok >= M) return;
  const float* xp = h + (size_t)tok * D;
  float v[4], s = 0.f;
  #pragma unroll
  for (int i = 0; i < 4; ++i) { v[i] = xp[lane + 32 * i]; s += v[i]; }
  #pragma unroll
  for (int o = 16; o > 0; o >>= 1) s += __shfl_xor(s, o, 32);
  float mean = s * (1.f / D);
  float var = 0.f;
  #pragma unroll
  for (int i = 0; i < 4; ++i) { float t = v[i] - mean; var += t * t; }
  #pragma unroll
  for (int o = 16; o > 0; o >>= 1) var += __shfl_xor(var, o, 32);
  float inv = rsqrtf(var * (1.f / D) + 1e-5f);
  _Float16* op = out + (size_t)tok * D;
  #pragma unroll
  for (int i = 0; i < 4; ++i) {
    int d = lane + 32 * i;
    op[d] = (_Float16)((v[i] - mean) * inv * sc[d] + bi[d]);
  }
}

// ------ generic WMMA GEMM: C[M,N] = A[M,K] * W[N,K]^T + bias --------------
// 4 waves / block; block owns a 16x64 slab; 64xK weight slab staged in LDS
// via the Tensor Data Mover (TDM) when available.
#define OP_F16       0
#define OP_RELU_F16  1
#define OP_RESID_F32 2

__global__ __launch_bounds__(128)
void k_gemm(const _Float16* __restrict__ A, const _Float16* __restrict__ W,
            const float* __restrict__ bias, _Float16* __restrict__ out16,
            float* __restrict__ resid, int M, int N, int K, int op) {
  extern __shared__ __align__(16) _Float16 wlds[];    // [64][K]
  int lane = threadIdx.x & 31, wv = threadIdx.x >> 5;
  int half_ = lane >> 4, ln = lane & 15;
  int nblock = blockIdx.x, mtile = blockIdx.y;

  // ---- stage W[nblock*64 .. +64][0..K] into LDS ----
#if __has_builtin(__builtin_amdgcn_tensor_load_to_lds)
  if (threadIdx.x < 32) {
    unsigned long long ga =
        (unsigned long long)(const void*)(W + (size_t)nblock * 64 * K);
    // D# group0: count=1, lds_addr=0, global_addr, type=2 ("image")
    v4u g0 = { 1u, 0u, (unsigned)(ga & 0xFFFFFFFFu),
               (unsigned)((ga >> 32) & 0x1FFFFFFu) | (2u << 30) };
    // D# group1: data_size=2B; tensor_dim0=K, tensor_dim1=N;
    //            tile_dim0=K, tile_dim1=64; tensor_dim0_stride=K
    v8i g1 = { (int)0x00010000,
               (int)((unsigned)(K & 0xFFFF) << 16),
               (int)((unsigned)(N & 0xFFFF) << 16),
               (int)((unsigned)(K & 0xFFFF) << 16),
               64, K, 0, 0 };
    v4i g2 = { 0, 0, 0, 0 };
    v4i g3 = { 0, 0, 0, 0 };
#if defined(__clang_major__) && __clang_major__ >= 23
    v8i g4 = { 0, 0, 0, 0, 0, 0, 0, 0 };
    __builtin_amdgcn_tensor_load_to_lds(g0, g1, g2, g3, g4, 0);
#else
    __builtin_amdgcn_tensor_load_to_lds(g0, g1, g2, g3, 0);
#endif
    __builtin_amdgcn_s_wait_tensorcnt(0);
  }
  __syncthreads();
#else
  {
    const uint4* src = (const uint4*)(W + (size_t)nblock * 64 * K);
    uint4* dst = (uint4*)wlds;
    int total = 64 * K / 8;                 // uint4 = 8 halfs
    for (int t = threadIdx.x; t < total; t += 128) dst[t] = src[t];
    __syncthreads();
  }
#endif

  int arow = mtile * 16 + ln; if (arow >= M) arow = M - 1;
  int wrow = wv * 16 + ln;                  // local row inside LDS slab
  int col  = nblock * 64 + wv * 16 + ln;
  float binit = bias ? bias[col] : 0.f;
  v8f c;
  #pragma unroll
  for (int r = 0; r < 8; ++r) c[r] = binit;

  for (int k0 = 0; k0 < K; k0 += 32) {
    v16h a = load_fragA(A, arow, K, k0, half_);
    v16h b = load_fragB(wlds, wrow, K, k0, half_);   // ds_load_b128 x2
    c = wmma_f16(a, b, c);
  }

  #pragma unroll
  for (int r = 0; r < 8; ++r) {
    int row = mtile * 16 + r + 8 * half_;
    if (row < M) {
      float v = c[r];
      if (op == OP_RESID_F32) {
        resid[(size_t)row * N + col] += v;
      } else {
        if (op == OP_RELU_F16) v = v > 0.f ? v : 0.f;
        out16[(size_t)row * N + col] = (_Float16)v;
      }
    }
  }
}

// ---------------- flash attention: 1 wave per (b,h,qtile) -------------------
__global__ __launch_bounds__(32)
void k_attn(const _Float16* __restrict__ qkv, _Float16* __restrict__ att) {
  const int LD = 3 * D;                       // 384
  int lane = threadIdx.x, half_ = lane >> 4, ln = lane & 15;
  int qtile = blockIdx.x;
  int bh = blockIdx.y, b = bh >> 2, hh = bh & 3;
  const _Float16* base  = qkv + (size_t)b * SS * LD;
  const _Float16* vbase = base + 2 * D + hh * DH;

  int qrow = qtile * 16 + ln; if (qrow >= SS) qrow = SS - 1;
  v16h qf = load_fragA(base, qrow, LD, hh * DH, half_);

  float m_st[8], l_st[8];
  v8f o0, o1;
  #pragma unroll
  for (int r = 0; r < 8; ++r) { m_st[r] = -1e30f; l_st[r] = 0.f; o0[r] = 0.f; o1[r] = 0.f; }
  const float scale = 0.17677669529663687f;   // 1/sqrt(32)

  __shared__ __align__(16) _Float16 plds[16 * 32];
  __shared__ __align__(16) _Float16 vlds[32 * 32];

  int nkb = (SS + 31) / 32;
  for (int jb = 0; jb < nkb; ++jb) {
    int j = jb * 32;

    // cooperative stage of V tile: lane -> one key row, 32 halfs (4x b128)
    {
      int vr = j + lane; if (vr >= SS) vr = SS - 1;
      const uint4* src = (const uint4*)(vbase + (size_t)vr * LD);
      uint4* dst = (uint4*)(vlds + lane * 32);
      dst[0] = src[0]; dst[1] = src[1]; dst[2] = src[2]; dst[3] = src[3];
    }

    int kr0 = j + ln;      if (kr0 >= SS) kr0 = SS - 1;
    int kr1 = j + 16 + ln; if (kr1 >= SS) kr1 = SS - 1;
    v16h kb0 = load_fragB(base, kr0, LD, D + hh * DH, half_);
    v16h kb1 = load_fragB(base, kr1, LD, D + hh * DH, half_);
    v8f z;
    #pragma unroll
    for (int r = 0; r < 8; ++r) z[r] = 0.f;
    v8f s0 = wmma_f16(qf, kb0, z);
    v8f s1 = wmma_f16(qf, kb1, z);

    bool v0 = (j + ln) < SS, v1 = (j + 16 + ln) < SS;
    float p0[8], p1[8];
    #pragma unroll
    for (int r = 0; r < 8; ++r) {
      float a0 = v0 ? s0[r] * scale : -1e30f;
      float a1 = v1 ? s1[r] * scale : -1e30f;
      float t = fmaxf(a0, a1);
      #pragma unroll
      for (int o = 8; o > 0; o >>= 1) t = fmaxf(t, __shfl_xor(t, o, 32));
      float mn = fmaxf(m_st[r], t);
      p0[r] = __expf(a0 - mn);
      p1[r] = __expf(a1 - mn);
      float ps = p0[r] + p1[r];
      #pragma unroll
      for (int o = 8; o > 0; o >>= 1) ps += __shfl_xor(ps, o, 32);
      float corr = __expf(m_st[r] - mn);
      l_st[r] = l_st[r] * corr + ps;
      o0[r] *= corr; o1[r] *= corr;
      m_st[r] = mn;
    }

    // stage P (C-layout) into LDS row-major, reload as A-fragment
    #pragma unroll
    for (int r = 0; r < 8; ++r) {
      int row = r + 8 * half_;
      plds[row * 32 + ln]      = (_Float16)p0[r];
      plds[row * 32 + 16 + ln] = (_Float16)p1[r];
    }
    asm volatile("s_wait_dscnt 0" ::: "memory");
    v16h pf = load_fragA(plds, ln, 32, 0, half_);

    // V fragments gathered (transposed) from LDS tile
    v16h vb0, vb1;
    #pragma unroll
    for (int e = 0; e < 16; ++e) {
      int kk = half_ * 16 + e;
      vb0[e] = vlds[kk * 32 + ln];
      vb1[e] = vlds[kk * 32 + 16 + ln];
    }
    o0 = wmma_f16(pf, vb0, o0);
    o1 = wmma_f16(pf, vb1, o1);
  }

  #pragma unroll
  for (int r = 0; r < 8; ++r) {
    int row = qtile * 16 + r + 8 * half_;
    if (row < SS) {
      float inv = 1.f / l_st[r];
      size_t o = ((size_t)(b * SS + row)) * D + hh * DH;
      att[o + ln]      = (_Float16)(o0[r] * inv);
      att[o + 16 + ln] = (_Float16)(o1[r] * inv);
    }
  }
}

// ---------------- head: LN(h[:,0]) . head_w + head_b ----------------
__global__ __launch_bounds__(128)
void k_head(const float* __restrict__ h, const float* __restrict__ sc,
            const float* __restrict__ bi, const float* __restrict__ hw,
            const float* __restrict__ hb, float* __restrict__ out) {
  __shared__ float red[128];
  int b = blockIdx.x, t = threadIdx.x;
  float x = h[(size_t)b * SS * D + t];
  red[t] = x; __syncthreads();
  for (int o = 64; o > 0; o >>= 1) { if (t < o) red[t] += red[t + o]; __syncthreads(); }
  float mean = red[0] * (1.f / D); __syncthreads();
  float dlt = x - mean;
  red[t] = dlt * dlt; __syncthreads();
  for (int o = 64; o > 0; o >>= 1) { if (t < o) red[t] += red[t + o]; __syncthreads(); }
  float inv = rsqrtf(red[0] * (1.f / D) + 1e-5f); __syncthreads();
  float y = dlt * inv * sc[t] + bi[t];
  red[t] = y * hw[t]; __syncthreads();
  for (int o = 64; o > 0; o >>= 1) { if (t < o) red[t] += red[t + o]; __syncthreads(); }
  if (t == 0) out[b] = red[0] + hb[0];
}

// ---------------- host launcher ----------------
extern "C" void kernel_launch(void* const* d_in, const int* in_sizes, int n_in,
                              void* d_out, int out_size, void* d_ws, size_t ws_size,
                              hipStream_t stream) {
  const float* x      = (const float*)d_in[0];
  const float* ew     = (const float*)d_in[1];
  const float* eb     = (const float*)d_in[2];
  const float* pos    = (const float*)d_in[3];
  const float* cls    = (const float*)d_in[4];
  const float* ipw    = (const float*)d_in[5];
  const float* ipb    = (const float*)d_in[6];
  const float* ow     = (const float*)d_in[7];
  const float* ob     = (const float*)d_in[8];
  const float* ln1s   = (const float*)d_in[9];
  const float* ln1b   = (const float*)d_in[10];
  const float* ln2s   = (const float*)d_in[11];
  const float* ln2b   = (const float*)d_in[12];
  const float* f1w    = (const float*)d_in[13];
  const float* f1b    = (const float*)d_in[14];
  const float* f2w    = (const float*)d_in[15];
  const float* f2b    = (const float*)d_in[16];
  const float* hlns   = (const float*)d_in[17];
  const float* hlnb   = (const float*)d_in[18];
  const float* hw     = (const float*)d_in[19];
  const float* hb     = (const float*)d_in[20];
  float* out = (float*)d_out;

  char* w = (char*)d_ws;
  auto carve = [&](size_t bytes) { char* p = w; w += (bytes + 255) & ~(size_t)255; return p; };
  float*     h     = (float*)    carve((size_t)MM * D * 4);
  _Float16*  hn16  = (_Float16*) carve((size_t)MM * D * 2);
  _Float16*  qkv16 = (_Float16*) carve((size_t)MM * 3 * D * 2);
  _Float16*  att16 = (_Float16*) carve((size_t)MM * D * 2);
  _Float16*  ff16  = (_Float16*) carve((size_t)MM * DFF * 2);
  _Float16*  ipw16 = (_Float16*) carve((size_t)NL * 3 * D * D * 2);
  _Float16*  ow16  = (_Float16*) carve((size_t)NL * D * D * 2);
  _Float16*  f1w16 = (_Float16*) carve((size_t)NL * DFF * D * 2);
  _Float16*  f2w16 = (_Float16*) carve((size_t)NL * D * DFF * 2);

  auto cvt = [&](const float* s, _Float16* d, int n) {
    k_cvt_f16<<<(n + 255) / 256, 256, 0, stream>>>(s, d, n);
  };
  cvt(ipw, ipw16, NL * 3 * D * D);
  cvt(ow,  ow16,  NL * D * D);
  cvt(f1w, f1w16, NL * DFF * D);
  cvt(f2w, f2w16, NL * D * DFF);

  k_embed<<<(MM * D + 255) / 256, 256, 0, stream>>>(x, ew, eb, pos, cls, h);

  const int MT = (MM + 15) / 16;          // 257 row tiles
  const int LNB = (MM + 3) / 4;           // ln blocks (4 tokens/block)

  for (int i = 0; i < NL; ++i) {
    k_ln_f16<<<LNB, 128, 0, stream>>>(h, ln1s + i * D, ln1b + i * D, hn16, MM);
    // qkv = hn @ in_proj_w^T + b   [MM, 384] f16
    k_gemm<<<dim3(3 * D / 64, MT), 128, 64 * D * 2, stream>>>(
        hn16, ipw16 + (size_t)i * 3 * D * D, ipb + i * 3 * D,
        qkv16, nullptr, MM, 3 * D, D, OP_F16);
    // attention -> att16 [MM, 128] f16
    k_attn<<<dim3((SS + 15) / 16, BB * H), 32, 0, stream>>>(qkv16, att16);
    // h += att @ out_w^T + b
    k_gemm<<<dim3(D / 64, MT), 128, 64 * D * 2, stream>>>(
        att16, ow16 + (size_t)i * D * D, ob + i * D,
        nullptr, h, MM, D, D, OP_RESID_F32);
    k_ln_f16<<<LNB, 128, 0, stream>>>(h, ln2s + i * D, ln2b + i * D, hn16, MM);
    // ff = relu(hn @ ff1_w^T + b) [MM, 512] f16
    k_gemm<<<dim3(DFF / 64, MT), 128, 64 * D * 2, stream>>>(
        hn16, f1w16 + (size_t)i * DFF * D, f1b + i * DFF,
        ff16, nullptr, MM, DFF, D, OP_RELU_F16);
    // h += ff @ ff2_w^T + b   (K = 512 -> 64 KB LDS slab)
    k_gemm<<<dim3(D / 64, MT), 128, 64 * DFF * 2, stream>>>(
        ff16, f2w16 + (size_t)i * D * DFF, f2b + i * D,
        nullptr, h, MM, D, DFF, OP_RESID_F32);
  }

  k_head<<<BB, 128, 0, stream>>>(h, hlns, hlnb, hw, hb, out);
}
